// s4_58720792870929
// MI455X (gfx1250) — matrix operand order repaired and checked by reading, not compiled, not run on hardware
//
#include <hip/hip_runtime.h>
#include <math.h>

// ---------------- problem constants ----------------
#define B_   16
#define NN   1024
#define T_   12
#define H_   128
#define S2_  32
#define NL_  2
#define BN_  (B_*NN)        // 16384 sequences
#define SEQ_PB 8            // sequences per block
#define NCOLS (SEQ_PB*T_)   // 96 GEMM columns per block
#define NTILES (NCOLS/16)   // 6 N-tiles of 16
#define YSTR 136            // padded K-stride (halves) per column: 272B = 17*16B

typedef __attribute__((ext_vector_type(16))) _Float16 v16h;
typedef __attribute__((ext_vector_type(8)))  _Float16 v8h;
typedef __attribute__((ext_vector_type(8)))  float    v8f;

// ---------------------------------------------------------------------------
// Kernel 1: build S4D convolution kernels k[l][c][t] (tiny, param-only).
// ---------------------------------------------------------------------------
__global__ __launch_bounds__(256)
void s4d_build_kernels(const float* __restrict__ logA,
                       const float* __restrict__ Aim,
                       const float* __restrict__ Cre,
                       const float* __restrict__ Cim,
                       const float* __restrict__ logdt,
                       float* __restrict__ kout) {
  int tid = threadIdx.x;          // 256 = NL*H
  int l = tid >> 7, c = tid & (H_ - 1);
  float dt = expf(logdt[l*H_ + c]);
  float acc[T_];
  #pragma unroll
  for (int t = 0; t < T_; ++t) acc[t] = 0.f;
  int base = (l*H_ + c)*S2_;
  for (int n = 0; n < S2_; ++n) {
    float ar = -expf(logA[base + n]);
    float ai = Aim[base + n];
    float dr = ar*dt, di = ai*dt;
    float er = expf(dr);
    float ec = er*cosf(di), es = er*sinf(di);     // e = exp(dtA)
    float mr = ec - 1.f,    mi = es;              // e - 1
    float cr = Cre[base + n], ci = Cim[base + n];
    float pr = cr*mr - ci*mi, pi = cr*mi + ci*mr; // C*(e-1)
    float inv = 1.f/(ar*ar + ai*ai);
    float qr = (pr*ar + pi*ai)*inv;               // /A
    float qi = (pi*ar - pr*ai)*inv;
    float wr = 1.f, wi = 0.f;                     // exp(dtA*t)
    #pragma unroll
    for (int t = 0; t < T_; ++t) {
      acc[t] += qr*wr - qi*wi;
      float nwr = wr*ec - wi*es;
      float nwi = wr*es + wi*ec;
      wr = nwr; wi = nwi;
    }
  }
  #pragma unroll
  for (int t = 0; t < T_; ++t) kout[(l*H_ + c)*T_ + t] = 2.f*acc[t];
}

// ---------------------------------------------------------------------------
// Kernel 2: fused encoder -> [conv+GELU -> GEMM(WMMA f16)+GLU+residual -> LN]x2
//           -> output + 1x1 skip. One block = 8 sequences, LDS-resident.
// LDS map: xb[96] f32 | hB[8*128*12] f32 | yB[96 cols * 136] f16 (K-contig,
//          padded) | LN scratch 576 f32.
// ---------------------------------------------------------------------------
__global__ __launch_bounds__(256)
void s4_fused(const float* __restrict__ x,
              const float* __restrict__ encW, const float* __restrict__ encB,
              const float* __restrict__ kconv, const float* __restrict__ Dp,
              const float* __restrict__ outW,  const float* __restrict__ outB,
              const float* __restrict__ lng,   const float* __restrict__ lnb,
              const float* __restrict__ scW,   const float* __restrict__ scB,
              float* __restrict__ out) {
  extern __shared__ char smem_raw[];
  float*    xb = (float*)smem_raw;                   // 96 f32
  float*    hB = xb + NCOLS;                         // 12288 f32 (48 KB)
  _Float16* yB = (_Float16*)(hB + SEQ_PB*H_*T_);     // 96*136 f16 (25.5 KB)
  float*    scrS = (float*)(yB + NCOLS*YSTR);        // 192: partial sums
  float*    scrQ = scrS + 2*NCOLS;                   // 192: partial sumsq
  float*    scrM = scrQ + 2*NCOLS;                   // 96: mu
  float*    scrR = scrM + NCOLS;                     // 96: rstd

  const int tid   = threadIdx.x;
  const int lane  = tid & 31;
  const int wave  = tid >> 5;          // 0..7
  const int laneM = lane & 15;
  const int half  = lane >> 4;
  const int seq0  = blockIdx.x * SEQ_PB;

  // ---- stage x for this block's 8 sequences (Cin == 1) ----
  if (tid < NCOLS) xb[tid] = x[seq0*T_ + tid];
  __syncthreads();

  // ---- encoder: h[s][c][t] = x[s,t]*enc_W[c] + enc_b[c] ----
  for (int i = tid; i < SEQ_PB*H_*T_; i += 256) {
    int t = i % T_;
    int c = (i / T_) & (H_ - 1);
    int s = i / (T_*H_);
    hB[i] = xb[s*T_ + t]*encW[c] + encB[c];
  }
  __syncthreads();

  for (int l = 0; l < NL_; ++l) {
    // ---- A fragments (out_W[l], f16). ISA 16-bit A 16x32 layout:
    // elements 0..7  <-> K = kk*32 + 8*half + e
    // elements 8..15 <-> K = kk*32 + 16 + 8*half + (e-8)
    // => two contiguous 8-float (2x float4) runs per fragment.
    v16h a0[4], a1[4];
    {
      const float* W = outW + (size_t)l*2*H_*H_;
      int m0 = wave*16 + laneM;        // channels 0..127   (GLU 'a')
      int m1 = 128 + m0;               // channels 128..255 (GLU 'b')
      #pragma unroll
      for (int kk = 0; kk < 4; ++kk) {
        const float4* p0 = (const float4*)(W + m0*H_ + kk*32 + 8*half);
        const float4* p1 = (const float4*)(W + m1*H_ + kk*32 + 8*half);
        float4 w0a = p0[0], w0b = p0[1];   // K base .. base+7
        float4 w0c = p0[4], w0d = p0[5];   // K base+16 .. base+23
        float4 w1a = p1[0], w1b = p1[1];
        float4 w1c = p1[4], w1d = p1[5];
        a0[kk] = (v16h){(_Float16)w0a.x,(_Float16)w0a.y,(_Float16)w0a.z,(_Float16)w0a.w,
                        (_Float16)w0b.x,(_Float16)w0b.y,(_Float16)w0b.z,(_Float16)w0b.w,
                        (_Float16)w0c.x,(_Float16)w0c.y,(_Float16)w0c.z,(_Float16)w0c.w,
                        (_Float16)w0d.x,(_Float16)w0d.y,(_Float16)w0d.z,(_Float16)w0d.w};
        a1[kk] = (v16h){(_Float16)w1a.x,(_Float16)w1a.y,(_Float16)w1a.z,(_Float16)w1a.w,
                        (_Float16)w1b.x,(_Float16)w1b.y,(_Float16)w1b.z,(_Float16)w1b.w,
                        (_Float16)w1c.x,(_Float16)w1c.y,(_Float16)w1c.z,(_Float16)w1c.w,
                        (_Float16)w1d.x,(_Float16)w1d.y,(_Float16)w1d.z,(_Float16)w1d.w};
      }
    }

    // ---- causal conv (T=12) + D*z + exact GELU -> yB[col][K] (f16) ----
    for (int p = tid; p < SEQ_PB*H_; p += 256) {
      int c = p & (H_ - 1), s = p >> 7;
      const float* kc = kconv + (l*H_ + c)*T_;
      float Dc = Dp[l*H_ + c];
      float z[T_], kv[T_];
      #pragma unroll
      for (int t = 0; t < T_; ++t) { z[t] = hB[(s*H_ + c)*T_ + t]; kv[t] = kc[t]; }
      #pragma unroll
      for (int t = 0; t < T_; ++t) {
        float acc = Dc*z[t];
        #pragma unroll
        for (int ss = 0; ss <= t; ++ss) acc += kv[ss]*z[t - ss];
        float g = 0.5f*acc*(1.f + erff(acc*0.70710678118654752f)); // exact GELU
        yB[(s*T_ + t)*YSTR + c] = (_Float16)g;                      // K-contig
      }
    }
    __syncthreads();

    // ---- GEMM (256x128)@(128x96) via v_wmma_f32_16x16x32_f16 ----
    // B 32x16 layout: lane(N=laneM,half): element e<8 <-> K = kk*32+8*half+e,
    // e>=8 <-> K = kk*32+16+8*half+(e-8) => 2x ds_load_b128.
    for (int nt = 0; nt < NTILES; ++nt) {
      int col = nt*16 + laneM;
      const _Float16* ycol = yB + col*YSTR + 8*half;
      v16h bfr[4];
      #pragma unroll
      for (int kk = 0; kk < 4; ++kk) {
        v8h lo = *(const v8h*)(ycol + kk*32);
        v8h hi = *(const v8h*)(ycol + kk*32 + 16);
        bfr[kk] = __builtin_shufflevector(lo, hi, 0,1,2,3,4,5,6,7,
                                                  8,9,10,11,12,13,14,15);
      }
      v8f acc0 = {}; v8f acc1 = {};
      #pragma unroll
      for (int kk = 0; kk < 4; ++kk) {
        acc0 = __builtin_amdgcn_wmma_f32_16x16x32_f16(false, a0[kk], false, bfr[kk],
                                                      (short)0, acc0, false, false);
        acc1 = __builtin_amdgcn_wmma_f32_16x16x32_f16(false, a1[kk], false, bfr[kk],
                                                      (short)0, acc1, false, false);
      }
      // D layout: element r <-> row m0 + r + 8*half, col = nt*16 + laneM
      int s = col / T_, t = col - s*T_;
      #pragma unroll
      for (int r = 0; r < 8; ++r) {
        int c0 = wave*16 + r + 8*half;
        float oa = acc0[r] + outB[l*2*H_ + c0];
        float ob = acc1[r] + outB[l*2*H_ + 128 + c0];
        // sigmoid via v_exp + v_rcp (avoids IEEE divide expansion)
        float g  = oa * __builtin_amdgcn_rcpf(1.f + __expf(-ob));   // GLU
        hB[(s*H_ + c0)*T_ + t] += g;                  // residual
      }
    }
    __syncthreads();

    // ---- post-norm LayerNorm over H per (seq,t) column ----
    // Pass A: 2 threads/column accumulate 64 channels each.
    if (tid < 2*NCOLS) {
      int col = tid >> 1, hf = tid & 1;
      int s = col / T_, t = col - s*T_;
      int cbase = hf*64;
      float su = 0.f, sq = 0.f;
      #pragma unroll 4
      for (int c = 0; c < 64; ++c) {
        float v = hB[(s*H_ + cbase + c)*T_ + t];
        su += v; sq += v*v;
      }
      scrS[tid] = su; scrQ[tid] = sq;
    }
    __syncthreads();
    // Pass B: finalize mu / rstd per column.
    if (tid < NCOLS) {
      float su = scrS[2*tid] + scrS[2*tid+1];
      float sq = scrQ[2*tid] + scrQ[2*tid+1];
      float mu = su*(1.f/H_);
      float var = sq*(1.f/H_) - mu*mu;
      scrM[tid] = mu;
      scrR[tid] = rsqrtf(var + 1e-5f);
    }
    __syncthreads();
    // Pass C: normalize, all 256 threads.
    for (int i = tid; i < SEQ_PB*H_*T_; i += 256) {
      int t = i % T_;
      int c = (i / T_) & (H_ - 1);
      int s = i / (T_*H_);
      int col = s*T_ + t;
      float v = hB[i];
      hB[i] = (v - scrM[col])*scrR[col]*lng[l*H_ + c] + lnb[l*H_ + c];
    }
    __syncthreads();
  }

  // ---- output (B,H,N,T) + 1x1 conv skip on original x ----
  for (int p = tid; p < SEQ_PB*H_; p += 256) {
    int c = p & (H_ - 1), s = p >> 7;
    int seq = seq0 + s;
    int b = seq >> 10, n = seq & (NN - 1);
    float sw = scW[c], sb = scB[c];
    size_t obase = (((size_t)b*H_ + c)*NN + n)*(size_t)T_;
    #pragma unroll
    for (int t = 0; t < T_; ++t)
      out[obase + t] = hB[(s*H_ + c)*T_ + t] + sw*xb[s*T_ + t] + sb;
  }
}

// ---------------------------------------------------------------------------
extern "C" void kernel_launch(void* const* d_in, const int* in_sizes, int n_in,
                              void* d_out, int out_size, void* d_ws, size_t ws_size,
                              hipStream_t stream) {
  (void)in_sizes; (void)n_in; (void)out_size; (void)ws_size;
  const float* x     = (const float*)d_in[0];
  const float* encW  = (const float*)d_in[1];
  const float* encB  = (const float*)d_in[2];
  const float* logA  = (const float*)d_in[3];
  const float* Aim   = (const float*)d_in[4];
  const float* Cre   = (const float*)d_in[5];
  const float* Cim   = (const float*)d_in[6];
  const float* logdt = (const float*)d_in[7];
  const float* Dp    = (const float*)d_in[8];
  const float* outW  = (const float*)d_in[9];
  const float* outB  = (const float*)d_in[10];
  const float* lng   = (const float*)d_in[11];
  const float* lnb   = (const float*)d_in[12];
  const float* scW   = (const float*)d_in[13];
  const float* scB   = (const float*)d_in[14];
  float* kws = (float*)d_ws;   // NL*H*T = 3072 floats

  s4d_build_kernels<<<1, 256, 0, stream>>>(logA, Aim, Cre, Cim, logdt, kws);

  size_t shmem = NCOLS*sizeof(float)                       // xb
               + (size_t)SEQ_PB*H_*T_*sizeof(float)        // hB
               + (size_t)NCOLS*YSTR*sizeof(_Float16)       // yB (padded)
               + (size_t)(2*NCOLS + 2*NCOLS + NCOLS + NCOLS)*sizeof(float); // LN scratch
  s4_fused<<<BN_/SEQ_PB, 256, shmem, stream>>>(x, encW, encB, kws, Dp,
                                               outW, outB, lng, lnb,
                                               scW, scB, (float*)d_out);
}
